// DualModalExpertContainer_85933705658887
// MI455X (gfx1250) — matrix operand order repaired and checked by reading, not compiled, not run on hardware
//
#include <hip/hip_runtime.h>
#include <hip/hip_bf16.h>

typedef __bf16 bf16_t;
typedef bf16_t v16bf __attribute__((ext_vector_type(16)));
typedef bf16_t v8bf  __attribute__((ext_vector_type(8)));
typedef float  v8f   __attribute__((ext_vector_type(8)));

#define HH 64
#define WW 64
#define HWP (HH*WW)
#define ASTR 40   // LDS activation row stride (bf16 units): 80B, 16B-aligned
#define WSTR 40   // LDS weight row stride

__device__ __forceinline__ float silu_f(float v){ return v / (1.0f + __expf(-v)); }

// ---------------- prep kernels ----------------
__global__ void cvt_f32_bf16(const float* __restrict__ in, bf16_t* __restrict__ out, int n){
    int i = blockIdx.x*256 + threadIdx.x;
    if(i < n) out[i] = (bf16_t)in[i];
}

__global__ void bn_fold(const float* __restrict__ g, const float* __restrict__ b,
                        const float* __restrict__ m, const float* __restrict__ v,
                        float* __restrict__ sb, int n){
    int i = blockIdx.x*64 + threadIdx.x;
    if(i < n){
        float s = g[i]*rsqrtf(v[i] + 1e-5f);
        sb[i]   = s;
        sb[n+i] = b[i] - m[i]*s;
    }
}

// ---------------- depthwise 5x5 (pad 2), optional fused BN+SiLU ----------------
__global__ __launch_bounds__(256) void dw5x5(
    const float* __restrict__ in, int CinTotal, int c0, int C,
    const float* __restrict__ w, const float* __restrict__ bnsb, int fuse,
    float* __restrict__ out)
{
    int t  = blockIdx.x*256 + threadIdx.x;        // over B*C*HW
    int px = t & (HWP-1);
    int bc = t >> 12;
    int c  = bc % C;
    int b  = bc / C;
    int y  = px >> 6, x = px & 63;
    const float* ip = in + ((size_t)(b*CinTotal + c0 + c))*HWP;
    const float* wp = w + c*25;
    float acc = 0.f;
    #pragma unroll
    for(int dy=-2; dy<=2; ++dy){
        int yy = y + dy;
        if((unsigned)yy >= HH) continue;
        #pragma unroll
        for(int dx=-2; dx<=2; ++dx){
            int xx = x + dx;
            if((unsigned)xx >= WW) continue;
            acc += ip[yy*WW + xx] * wp[(dy+2)*5 + (dx+2)];
        }
    }
    if(fuse){
        float s = bnsb[c], bb = bnsb[C + c];
        acc = silu_f(acc*s + bb);
    }
    out[((size_t)(b*C + c))*HWP + px] = acc;
}

// ---------------- implicit-GEMM conv via WMMA bf16 ----------------
// Compile-time shapes; double-buffered LDS; software-pipelined staging:
//   global loads (next tile) -> WMMA block (current tile) -> cvt+LDS stores (next tile)
// Staging: thread owns 16 consecutive k for one pixel; OOB via clamped loads + mask
// select (no exec-mask branches); LDS writes as 2x ds_store_b128.
template<int CIN, int KS, int PAD, int COUT>
__global__ __launch_bounds__(256) void convgemm_wmma(
    const float* __restrict__ in, int CinTotal, int c0,
    const bf16_t* __restrict__ wpack, const float* __restrict__ bnsb,
    float* __restrict__ out)
{
    constexpr int KS2  = KS*KS;
    constexpr int KTOT = CIN*KS2;
    constexpr int NKT  = KTOT >> 5;           // K tiles of 32
    constexpr int NCT  = COUT >> 4;           // channel tiles (4 or 8)

    __shared__ __align__(16) bf16_t ldsA[2][128*ASTR];   // activations: [pixel][k]
    __shared__ __align__(16) bf16_t ldsW[2][COUT*WSTR];  // weights:     [cout ][k]

    const int t    = threadIdx.x;
    const int lane = t & 31;
    const int wv   = t >> 5;                  // wave id = pixel tile

    const int blockPix = blockIdx.x * 128;    // all 128 pixels in same image (HW=4096)
    const int b    = blockPix >> 12;
    const int pix0 = blockPix & (HWP-1);

    v8f acc[NCT];
    #pragma unroll
    for(int i=0;i<NCT;++i)
        #pragma unroll
        for(int j=0;j<8;++j) acc[i][j] = 0.f;

    const int n  = lane & 15;
    const int kg = lane >> 4;

    const float* inb = in + (size_t)(b*CinTotal + c0)*HWP;

    // staging mapping: thread owns pixel p, 16 consecutive k at offset koff
    const int p    = t & 127;
    const int hi   = t >> 7;        // 0 or 1
    const int koff = hi << 4;       // 0 or 16
    const int P    = pix0 + p;
    const int py   = P >> 6;
    const int pxc  = P & 63;
    const int wrow = t >> 1, whalf = t & 1;   // weight staging: 32B per thread

    float aReg[16];
    unsigned amask = 0;
    uint4 wr0, wr1;

    // ---- load phase: global -> registers ----
    auto loadTile = [&](int kt){
        const int kbase = kt*32;
        amask = 0;
        #pragma unroll
        for(int j=0; j<16; ++j){
            if constexpr (KS == 1){
                int ci = kbase + koff + j;
                aReg[j] = inb[(size_t)ci*HWP + P];
            } else {
                // uniform decomposition of the two candidate k values (SALU) + select
                int k0  = kbase + j;         // koff==0 candidate (uniform)
                int k1  = k0 + 16;           // koff==16 candidate (uniform)
                int ci0 = k0/KS2,      r0 = k0 - ci0*KS2;
                int dy0 = r0/KS - PAD, dx0 = r0 - (r0/KS)*KS - PAD;
                int ci1 = k1/KS2,      r1 = k1 - ci1*KS2;
                int dy1 = r1/KS - PAD, dx1 = r1 - (r1/KS)*KS - PAD;
                int ci = hi ? ci1 : ci0;
                int dy = hi ? dy1 : dy0;
                int dx = hi ? dx1 : dx0;
                int yy = py  + dy;
                int xx = pxc + dx;
                bool ok = ((unsigned)yy < HH) && ((unsigned)xx < WW);
                int yc = min(max(yy, 0), HH-1);
                int xc = min(max(xx, 0), WW-1);
                aReg[j] = inb[(size_t)ci*HWP + yc*WW + xc];   // clamped, always valid
                amask |= ok ? (1u << j) : 0u;
            }
        }
        if(wrow < COUT){
            const uint4* s4 = (const uint4*)(wpack + (size_t)wrow*KTOT + kbase + whalf*16);
            wr0 = s4[0];
            wr1 = s4[1];
        }
    };

    // ---- store phase: registers -> LDS (bf16, packed 128-bit stores) ----
    auto storeTile = [&](int buf){
        v8bf t0, t1;
        #pragma unroll
        for(int j=0; j<8; ++j){
            float v0 = aReg[j],   v1 = aReg[j+8];
            if constexpr (KS != 1){
                v0 = ((amask >> j)     & 1u) ? v0 : 0.f;
                v1 = ((amask >> (j+8)) & 1u) ? v1 : 0.f;
            }
            t0[j] = (bf16_t)v0;
            t1[j] = (bf16_t)v1;
        }
        *((v8bf*)&ldsA[buf][p*ASTR + koff    ]) = t0;
        *((v8bf*)&ldsA[buf][p*ASTR + koff + 8]) = t1;
        if(wrow < COUT){
            *((uint4*)&ldsW[buf][wrow*WSTR + whalf*16    ]) = wr0;
            *((uint4*)&ldsW[buf][wrow*WSTR + whalf*16 + 8]) = wr1;
        }
    };

    // prologue: stage tile 0 into buffer 0
    loadTile(0);
    storeTile(0);

    for(int kt=0; kt<NKT; ++kt){
        const int buf = kt & 1;
        __syncthreads();

        // issue next tile's global loads (latency hides under WMMA block)
        if(kt+1 < NKT) loadTile(kt+1);

        // ---- B fragment: this wave's 16-pixel tile, 16 contiguous K per lane ----
        v16bf bfrag;
        {
            const uint4* p0 = (const uint4*)&ldsA[buf][(wv*16 + n)*ASTR + kg*16];
            ((uint4*)&bfrag)[0] = p0[0];
            ((uint4*)&bfrag)[1] = p0[1];
        }
        // ---- A fragments per channel tile + WMMA ----
        #pragma unroll
        for(int ct=0; ct<NCT; ++ct){
            v16bf afrag;
            const uint4* q0 = (const uint4*)&ldsW[buf][(ct*16 + n)*WSTR + kg*16];
            const uint4* q1 = (const uint4*)&ldsW[buf][(ct*16 + n)*WSTR + kg*16 + 8];
            ((uint4*)&afrag)[0] = q0[0];
            ((uint4*)&afrag)[1] = q1[0];
            acc[ct] = __builtin_amdgcn_wmma_f32_16x16x32_bf16(
                false, afrag, false, bfrag, (short)0, acc[ct], false, false);
        }

        // convert + store next tile into the other buffer
        if(kt+1 < NKT) storeTile(buf ^ 1);
    }

    // ---- epilogue: BN + SiLU, pixel-coalesced stores ----
    const int mh  = (lane >> 4) * 8;          // M offset for upper half-wave
    const int opx = pix0 + wv*16 + n;
    #pragma unroll
    for(int ct=0; ct<NCT; ++ct){
        #pragma unroll
        for(int r=0; r<8; ++r){
            int ch = ct*16 + mh + r;
            float s  = bnsb[ch];
            float bb = bnsb[COUT + ch];
            float v  = silu_f(acc[ct][r]*s + bb);
            out[((size_t)(b*COUT + ch))*HWP + opx] = v;
        }
    }
}

// ---------------- routed mixture ----------------
__global__ __launch_bounds__(256) void mixture(
    const float* __restrict__ x,  const float* __restrict__ wts, const int* __restrict__ idx,
    const float* __restrict__ e0, const float* __restrict__ e1,
    const float* __restrict__ g1, const float* __restrict__ g2,
    float* __restrict__ out)
{
    size_t t = (size_t)blockIdx.x*256 + threadIdx.x;   // over 32*128*4096
    int px = (int)(t & (HWP-1));
    int bc = (int)(t >> 12);
    int c  = bc & 127;
    int b  = bc >> 7;
    float xv  = x[t];
    float e0v = e0[t];
    float e1v = e1[t];
    float e2v = (c < 64) ? g1[((size_t)(b*64 + c     ))*HWP + px]
                         : g2[((size_t)(b*64 + (c-64)))*HWP + px];
    float o = 0.f;
    #pragma unroll
    for(int k=0;k<2;++k){
        int   id = idx[b*2+k];
        float w  = wts[b*2+k];
        o += (id==0) ? e0v*w   : 0.f;
        o += (id==1) ? e1v*w   : 0.f;
        o += (id==2) ? e2v*w   : 0.f;
        o += (id==3) ? xv*0.1f : 0.f;
    }
    out[t] = o;
}

// ---------------- host launch ----------------
extern "C" void kernel_launch(void* const* d_in, const int* in_sizes, int n_in,
                              void* d_out, int out_size, void* d_ws, size_t ws_size,
                              hipStream_t stream)
{
    const float* x      = (const float*)d_in[0];
    const float* wts    = (const float*)d_in[1];
    const int*   idx    = (const int*)  d_in[2];
    const float* e0_w   = (const float*)d_in[3];
    const float* e0_g   = (const float*)d_in[4];
    const float* e0_b   = (const float*)d_in[5];
    const float* e0_m   = (const float*)d_in[6];
    const float* e0_v   = (const float*)d_in[7];
    const float* e1_dw  = (const float*)d_in[8];
    const float* e1_pw  = (const float*)d_in[9];
    const float* e1_g   = (const float*)d_in[10];
    const float* e1_b   = (const float*)d_in[11];
    const float* e1_m   = (const float*)d_in[12];
    const float* e1_v   = (const float*)d_in[13];
    const float* e2a_w  = (const float*)d_in[14];
    const float* e2a_g  = (const float*)d_in[15];
    const float* e2a_b  = (const float*)d_in[16];
    const float* e2a_m  = (const float*)d_in[17];
    const float* e2a_v  = (const float*)d_in[18];
    const float* e2b_w  = (const float*)d_in[19];
    const float* e2b_g  = (const float*)d_in[20];
    const float* e2b_b  = (const float*)d_in[21];
    const float* e2b_m  = (const float*)d_in[22];
    const float* e2b_v  = (const float*)d_in[23];
    float* out = (float*)d_out;
    char*  ws  = (char*)d_ws;

    const int B = 32, C = 128, CS = 64;
    const size_t big  = (size_t)B*C *HWP*sizeof(float);   // 67 MB
    const size_t half = (size_t)B*CS*HWP*sizeof(float);   // 33.5 MB

    size_t o = 0;
    auto alloc = [&](size_t bytes)->size_t { size_t r = o; o += (bytes + 255) & ~(size_t)255; return r; };
    size_t o_e0wb  = alloc((size_t)C*576*2);
    size_t o_e1wb  = alloc((size_t)C*64*2);
    size_t o_e2awb = alloc((size_t)CS*1152*2);
    size_t o_bn0   = alloc(2*C*4);
    size_t o_bn1   = alloc(2*C*4);
    size_t o_bn2a  = alloc(2*CS*4);
    size_t o_bn2b  = alloc(2*CS*4);
    size_t o_dw1   = alloc(half);
    size_t o_e0    = alloc(big);
    size_t o_e1    = alloc(big);
    size_t o_g1    = alloc(half);
    size_t o_g2    = alloc(half);
    (void)ws_size; (void)in_sizes; (void)n_in; (void)out_size;

    bf16_t* e0_wb  = (bf16_t*)(ws + o_e0wb);
    bf16_t* e1_wb  = (bf16_t*)(ws + o_e1wb);
    bf16_t* e2a_wb = (bf16_t*)(ws + o_e2awb);
    float*  bn0    = (float*)(ws + o_bn0);
    float*  bn1    = (float*)(ws + o_bn1);
    float*  bn2a   = (float*)(ws + o_bn2a);
    float*  bn2b   = (float*)(ws + o_bn2b);
    float*  dw1    = (float*)(ws + o_dw1);
    float*  e0buf  = (float*)(ws + o_e0);
    float*  e1buf  = (float*)(ws + o_e1);
    float*  g1buf  = (float*)(ws + o_g1);
    float*  g2buf  = (float*)(ws + o_g2);

    // weight packing (f32 -> bf16, already [Cout][K] contiguous in OIHW flatten)
    cvt_f32_bf16<<<(C*576  + 255)/256, 256, 0, stream>>>(e0_w,  e0_wb,  C*576);
    cvt_f32_bf16<<<(C*64   + 255)/256, 256, 0, stream>>>(e1_pw, e1_wb,  C*64);
    cvt_f32_bf16<<<(CS*1152+ 255)/256, 256, 0, stream>>>(e2a_w, e2a_wb, CS*1152);

    // BN folds
    bn_fold<<<2, 64, 0, stream>>>(e0_g,  e0_b,  e0_m,  e0_v,  bn0,  C);
    bn_fold<<<2, 64, 0, stream>>>(e1_g,  e1_b,  e1_m,  e1_v,  bn1,  C);
    bn_fold<<<1, 64, 0, stream>>>(e2a_g, e2a_b, e2a_m, e2a_v, bn2a, CS);
    bn_fold<<<1, 64, 0, stream>>>(e2b_g, e2b_b, e2b_m, e2b_v, bn2b, CS);

    const int dwBlocks = (B*CS*HWP)/256;   // 32768
    // e1 depthwise 5x5 on x_ir (no BN yet)
    dw5x5<<<dwBlocks, 256, 0, stream>>>(x, C, CS, CS, e1_dw, nullptr, 0, dw1);

    const int gemmBlocks = (B*HWP)/128;    // 1024
    // e0: conv3x3 x_rgb (64 -> 128) + BN + SiLU
    convgemm_wmma<64, 3, 1, 128><<<gemmBlocks, 256, 0, stream>>>(x,   C,  0, e0_wb,  bn0,  e0buf);
    // e1: pointwise 1x1 on dw1 (64 -> 128) + BN + SiLU
    convgemm_wmma<64, 1, 0, 128><<<gemmBlocks, 256, 0, stream>>>(dw1, CS, 0, e1_wb,  bn1,  e1buf);
    // g1: conv3x3 full x (128 -> 64) + BN + SiLU
    convgemm_wmma<128, 3, 1, 64><<<gemmBlocks, 256, 0, stream>>>(x,   C,  0, e2a_wb, bn2a, g1buf);
    // g2: depthwise 5x5 on g1 + BN + SiLU (fused)
    dw5x5<<<dwBlocks, 256, 0, stream>>>(g1buf, CS, 0, CS, e2b_w, bn2b, 1, g2buf);

    // routed mixture
    mixture<<<(B*C*HWP)/256, 256, 0, stream>>>(x, wts, idx, e0buf, e1buf, g1buf, g2buf, out);
}
